// EntityExtraction_69441031242560
// MI455X (gfx1250) — compile-verified, matrix operand order checked by reading, not run on hardware
//
#include <hip/hip_runtime.h>
#include <math.h>

// ---- problem constants (from reference) ----
#define BB 32
#define SS 256
#define LCC 16
#define WDIM 256
#define CDIM 124
#define CNNO 32
#define POSD 36
#define ENRD 7
#define HH 512
#define TT 18
#define IN0D 331
#define IN0P 352          // padded to multiple of 32
#define K0TOT 864         // 352 + 512
#define IN1D 1024
#define K1TOT 1536        // 1024 + 512
#define FOURH 2048
#define NROWS (BB*SS)     // 8192

typedef __attribute__((ext_vector_type(16))) __bf16 v16bf;
typedef __attribute__((ext_vector_type(8)))  __bf16 v8bf;
typedef __attribute__((ext_vector_type(8)))  float  v8f;
typedef __attribute__((ext_vector_type(4)))  int    v4i;

// CDNA5 async global<->LDS paths (guarded: fall back to sync copies if the
// toolchain doesn't declare the builtins).
#if __has_builtin(__builtin_amdgcn_global_load_async_to_lds_b128) && \
    __has_builtin(__builtin_amdgcn_s_wait_asynccnt)
#define HAVE_ASYNC_LD 1
#else
#define HAVE_ASYNC_LD 0
#endif
#if HAVE_ASYNC_LD && __has_builtin(__builtin_amdgcn_global_store_async_from_lds_b128)
#define HAVE_ASYNC_ST 1
#else
#define HAVE_ASYNC_ST 0
#endif

typedef __attribute__((address_space(1))) void* gas_void;
typedef __attribute__((address_space(3))) void* las_void;
typedef __attribute__((address_space(1))) v4i*  gas_v4i;
typedef __attribute__((address_space(3))) v4i*  las_v4i;

__device__ __forceinline__ gas_v4i as_global_v4i(const void* p) {
  return (gas_v4i)(gas_void)p;
}
__device__ __forceinline__ las_v4i as_lds_v4i(void* p) {
  return (las_v4i)(las_void)p;
}

// bf16 WMMA fragment load (A from LDS rows or B from row-major weights).
// Per ISA 7.12.2: lane holds row (lane&15); lanes<16 take K chunks {0..7,16..23},
// lanes>=16 take {8..15,24..31} of each 32-wide K slice -> two 16B loads.
__device__ __forceinline__ v16bf load_frag(const __bf16* rowbase, int kk, int lane) {
  const __bf16* p = rowbase + kk * 32 + ((lane & 16) ? 8 : 0);
  v8bf lo = *(const v8bf*)(p);
  v8bf hi = *(const v8bf*)(p + 16);
  v16bf r;
#pragma unroll
  for (int i = 0; i < 8; ++i) { r[i] = lo[i]; r[i + 8] = hi[i]; }
  return r;
}

__device__ __forceinline__ float sigm(float x) { return 1.f / (1.f + expf(-x)); }

// ---- weight prep: [Wih | pad | Whh] -> bf16 (2048 x ktot) ----
__global__ void pack_wcat_kernel(const float* __restrict__ Wih, const float* __restrict__ Whh,
                                 __bf16* __restrict__ dst, int din, int din_pad, int ktot) {
  int idx = blockIdx.x * blockDim.x + threadIdx.x;
  int total = FOURH * ktot;
  if (idx >= total) return;
  int n = idx / ktot, k = idx - n * ktot;
  float v = 0.f;
  if (k < din)                               v = Wih[(size_t)n * din + k];
  else if (k >= din_pad && k < din_pad + HH) v = Whh[(size_t)n * HH + (k - din_pad)];
  dst[idx] = (__bf16)v;
}

// ---- linear weight rows f32 -> bf16 with row padding ----
__global__ void cvt_rows_kernel(const float* __restrict__ src, __bf16* __restrict__ dst,
                                int N, int Npad, int K) {
  int idx = blockIdx.x * blockDim.x + threadIdx.x;
  if (idx >= Npad * K) return;
  int n = idx / K, k = idx - n * K;
  dst[idx] = (__bf16)((n < N) ? src[(size_t)n * K + k] : 0.f);
}

// ---- char CNN: one block per (b,s); ce staged in LDS ----
__global__ void char_cnn_kernel(const int* __restrict__ x_char, const float* __restrict__ cemb,
                                const float* __restrict__ cw, const float* __restrict__ cb,
                                float* __restrict__ cpool) {
  __shared__ float ce[LCC * CDIM];
  int idx = blockIdx.x;
  int tid = threadIdx.x;
  for (int i = tid; i < LCC * CDIM; i += 128) {
    int l = i / CDIM, c = i - l * CDIM;
    ce[i] = cemb[(size_t)x_char[idx * LCC + l] * CDIM + c];
  }
  __syncthreads();
  if (tid < CNNO) {
    float mx = -1e30f;
    for (int pos = 0; pos < LCC - 4; ++pos) {
      float acc = 0.f;
      for (int c = 0; c < CDIM; ++c) {
        const float* wrow = cw + (size_t)tid * CDIM * 5 + c * 5;
#pragma unroll
        for (int tp = 0; tp < 5; ++tp) acc += ce[(pos + tp) * CDIM + c] * wrow[tp];
      }
      mx = fmaxf(mx, acc);
    }
    cpool[idx * CNNO + tid] = mx + cb[tid];
  }
}

// ---- build relu(concat(...)) as bf16 rows (8192 x 352) ----
__global__ void build_x_kernel(const int* __restrict__ xw, const float* __restrict__ xp,
                               const float* __restrict__ xe, const float* __restrict__ wemb,
                               const float* __restrict__ cpool, __bf16* __restrict__ x0) {
  int idx = blockIdx.x;
  int tid = threadIdx.x;
  int w = xw[idx];
  for (int col = tid; col < IN0P; col += 128) {
    float v = 0.f;
    if (col < WDIM)                    v = wemb[(size_t)w * WDIM + col];
    else if (col < WDIM + POSD)        v = xp[(size_t)idx * POSD + (col - WDIM)];
    else if (col < WDIM + POSD + CNNO) v = cpool[idx * CNNO + (col - WDIM - POSD)];
    else if (col < IN0D)               v = xe[(size_t)idx * ENRD + (col - WDIM - POSD - CNNO)];
    v = fmaxf(v, 0.f);
    x0[(size_t)idx * IN0P + col] = (__bf16)v;
  }
}

// ---- persistent LSTM direction kernel: grid=2 (fwd/bwd), block=1024 (32 waves) ----
// LDS: xbuf0/xbuf1 (double-buffered x_t, 32 x din_pad each), hbuf (32 x 512),
//      z (32 x 2048). Cell state c lives in registers. x_{t+1} is prefetched
//      into the alternate buffer via async global->LDS while step t computes.
__global__ void lstm_layer_kernel(const __bf16* __restrict__ Xin, int xstride, int din_pad, int ktot,
                                  const __bf16* __restrict__ WcatF, const __bf16* __restrict__ WcatB,
                                  const float* __restrict__ bF, const float* __restrict__ bB,
                                  __bf16* __restrict__ Hout) {
  extern __shared__ __align__(16) char smem[];
  __bf16* xbuf0 = (__bf16*)smem;                       // 32 * din_pad
  __bf16* xbuf1 = xbuf0 + (size_t)BB * din_pad;        // 32 * din_pad
  __bf16* hbuf  = xbuf1 + (size_t)BB * din_pad;        // 32 * 512
  __bf16* z     = hbuf  + (size_t)BB * HH;             // 32 * 2048

  const int tid  = threadIdx.x;
  const int lane = tid & 31;
  const int wave = tid >> 5;
  const bool fwd = (blockIdx.x == 0);
  const __bf16* Wcat = fwd ? WcatF : WcatB;
  const float*  bias = fwd ? bF : bB;
  const int dirOff   = fwd ? 0 : HH;

  // zero h
  for (int c = tid; c < (BB * HH) / 8; c += 1024) {
    int b = c >> 6, j = (c & 63) << 3;
    *(uint4*)&hbuf[b * HH + j] = make_uint4(0u, 0u, 0u, 0u);
  }
  float creg[16];
#pragma unroll
  for (int q = 0; q < 16; ++q) creg[q] = 0.f;

  const int xch = din_pad >> 3;        // 16B chunks per input row
  const int xk  = din_pad >> 5;        // 32-wide K slices in x region

  // stage x for step 0 into xbuf0
  {
    int t0 = fwd ? 0 : (SS - 1);
    for (int c = tid; c < BB * xch; c += 1024) {
      int b = c / xch, off = (c - b * xch) << 3;
      const __bf16* src = &Xin[((size_t)(b * SS + t0)) * xstride + off];
      __bf16* dst = &xbuf0[b * din_pad + off];
#if HAVE_ASYNC_LD
      __builtin_amdgcn_global_load_async_to_lds_b128(as_global_v4i(src), as_lds_v4i(dst), 0, 0);
#else
      *(uint4*)dst = *(const uint4*)src;
#endif
    }
#if HAVE_ASYNC_LD
    __builtin_amdgcn_s_wait_asynccnt(0);
#endif
  }
  __syncthreads();

  for (int step = 0; step < SS; ++step) {
    int t = fwd ? step : (SS - 1 - step);
    __bf16* xcur = (step & 1) ? xbuf1 : xbuf0;
    __bf16* xnxt = (step & 1) ? xbuf0 : xbuf1;

    // prefetch x_{t+1} into the alternate buffer (async: overlaps the GEMM)
    if (step + 1 < SS) {
      int tn = fwd ? (step + 1) : (SS - 2 - step);
      for (int c = tid; c < BB * xch; c += 1024) {
        int b = c / xch, off = (c - b * xch) << 3;
        const __bf16* src = &Xin[((size_t)(b * SS + tn)) * xstride + off];
        __bf16* dst = &xnxt[b * din_pad + off];
#if HAVE_ASYNC_LD
        __builtin_amdgcn_global_load_async_to_lds_b128(as_global_v4i(src), as_lds_v4i(dst), 0, 0);
#else
        *(uint4*)dst = *(const uint4*)src;
#endif
      }
    }

    // z(32x2048) = [x_t ; h] @ Wcat^T via WMMA; 32 waves x 8 tiles cover 2x128 tiles
#pragma unroll
    for (int tile = 0; tile < 8; ++tile) {
      int m_tile = tile & 1;
      int n_tile = (wave << 2) + (tile >> 1);
      const __bf16* arowx = xcur + (size_t)(m_tile * 16 + (lane & 15)) * din_pad;
      const __bf16* arowh = hbuf + (size_t)(m_tile * 16 + (lane & 15)) * HH;
      const __bf16* brow  = Wcat + (size_t)(n_tile * 16 + (lane & 15)) * ktot;
      v8f acc = {0.f, 0.f, 0.f, 0.f, 0.f, 0.f, 0.f, 0.f};
      for (int kk = 0; kk < xk; ++kk) {              // x part
        __builtin_prefetch(brow + (kk + 1) * 32, 0, 1);   // global_prefetch_b8
        v16bf av = load_frag(arowx, kk, lane);
        v16bf bv = load_frag(brow, kk, lane);
        acc = __builtin_amdgcn_wmma_f32_16x16x32_bf16(false, av, false, bv,
                                                      (short)0, acc, false, false);
      }
      for (int kh = 0; kh < (HH >> 5); ++kh) {       // h part
        __builtin_prefetch(brow + (xk + kh + 1) * 32, 0, 1);
        v16bf av = load_frag(arowh, kh, lane);
        v16bf bv = load_frag(brow, xk + kh, lane);
        acc = __builtin_amdgcn_wmma_f32_16x16x32_bf16(false, av, false, bv,
                                                      (short)0, acc, false, false);
      }
#pragma unroll
      for (int r = 0; r < 8; ++r) {
        int m = m_tile * 16 + r + ((lane >> 4) << 3);
        int n = n_tile * 16 + (lane & 15);
        z[m * FOURH + n] = (__bf16)acc[r];
      }
    }
    __syncthreads();

    // gates: each thread owns two contiguous 8-wide (b, j0..j0+7) chunks
#pragma unroll
    for (int half = 0; half < 2; ++half) {
      int cc = half * 1024 + tid;
      int b = cc >> 6, j0 = (cc & 63) << 3;
#pragma unroll
      for (int e = 0; e < 8; ++e) {
        int j = j0 + e, q = half * 8 + e;
        float zi = (float)z[b * FOURH + j]        + bias[j];
        float zf = (float)z[b * FOURH + 512 + j]  + bias[512 + j];
        float zg = (float)z[b * FOURH + 1024 + j] + bias[1024 + j];
        float zo = (float)z[b * FOURH + 1536 + j] + bias[1536 + j];
        float c2 = sigm(zf) * creg[q] + sigm(zi) * tanhf(zg);
        float h2 = sigm(zo) * tanhf(c2);
        creg[q] = c2;
        __bf16 hb = (__bf16)h2;
        hbuf[b * HH + j] = hb;
#if !HAVE_ASYNC_ST
        Hout[((size_t)(b * SS + t)) * IN1D + dirOff + j] = hb;
#endif
      }
#if HAVE_ASYNC_ST
      // write the freshly produced 16B chunk of h straight from LDS to global
#if __has_builtin(__builtin_amdgcn_s_wait_dscnt)
      __builtin_amdgcn_s_wait_dscnt(0);   // ds stores above must land first
#endif
      __bf16* lsrc = &hbuf[b * HH + j0];
      __bf16* gdst = &Hout[((size_t)(b * SS + t)) * IN1D + dirOff + j0];
      __builtin_amdgcn_global_store_async_from_lds_b128(as_global_v4i(gdst), as_lds_v4i(lsrc), 0, 0);
#endif
    }

#if HAVE_ASYNC_LD
    __builtin_amdgcn_s_wait_asynccnt(0);   // x_{t+1} staged, h stores drained
#endif
    __syncthreads();
  }
}

// ---- generic WMMA GEMM + bias: one 16x16 tile per wave/block ----
__global__ void gemm_bias_kernel(const __bf16* __restrict__ A, int Ka,
                                 const __bf16* __restrict__ W, const float* __restrict__ bias,
                                 int N, float* __restrict__ Cf, __bf16* __restrict__ Cb, int ldc) {
  int lane = threadIdx.x & 31;
  int mt = blockIdx.x, nt = blockIdx.y;
  const __bf16* arow = A + (size_t)(mt * 16 + (lane & 15)) * Ka;
  const __bf16* brow = W + (size_t)(nt * 16 + (lane & 15)) * Ka;
  v8f acc = {0.f, 0.f, 0.f, 0.f, 0.f, 0.f, 0.f, 0.f};
  int kiters = Ka >> 5;
  for (int kk = 0; kk < kiters; ++kk) {
    v16bf av = load_frag(arow, kk, lane);
    v16bf bv = load_frag(brow, kk, lane);
    acc = __builtin_amdgcn_wmma_f32_16x16x32_bf16(false, av, false, bv,
                                                  (short)0, acc, false, false);
  }
#pragma unroll
  for (int r = 0; r < 8; ++r) {
    int m = mt * 16 + r + ((lane >> 4) << 3);
    int n = nt * 16 + (lane & 15);
    if (n < N) {
      float v = acc[r] + bias[n];
      if (Cf) Cf[(size_t)m * ldc + n] = v;
      if (Cb) Cb[(size_t)m * ldc + n] = (__bf16)v;
    }
  }
}

// ---- Viterbi: one block (32 threads) per batch row; scores in LDS ----
__global__ void viterbi_kernel(const float* __restrict__ em, const float* __restrict__ st,
                               const float* __restrict__ en, const float* __restrict__ tr,
                               int* __restrict__ hist, float* __restrict__ dec) {
  __shared__ float sc[32];
  __shared__ float trs[TT * TT];
  int b = blockIdx.x, j = threadIdx.x;
  for (int i = j; i < TT * TT; i += 32) trs[i] = tr[i];
  if (j < TT) sc[j] = st[j] + em[(size_t)(b * SS) * TT + j];
  __syncthreads();
  for (int s = 1; s < SS; ++s) {
    float best = -1e30f; int bi = 0;
    if (j < TT) {
      for (int i = 0; i < TT; ++i) {
        float v = sc[i] + trs[i * TT + j];
        if (v > best) { best = v; bi = i; }
      }
      best += em[(size_t)(b * SS + s) * TT + j];
    }
    __syncthreads();
    if (j < TT) { sc[j] = best; hist[(b * SS + s) * 32 + j] = bi; }
    __syncthreads();
  }
  if (j == 0) {
    float bv = -1e30f; int last = 0;
    for (int i = 0; i < TT; ++i) { float v = sc[i] + en[i]; if (v > bv) { bv = v; last = i; } }
    int tag = last;
    dec[b * SS + SS - 1] = (float)last;
    for (int s = SS - 1; s >= 1; --s) {
      tag = hist[(b * SS + s) * 32 + tag];
      dec[b * SS + s - 1] = (float)tag;
    }
  }
}

// ---- CRF forward (logZ) + numerator per batch row ----
__global__ void crf_nll_kernel(const float* __restrict__ em, const int* __restrict__ tags,
                               const unsigned char* __restrict__ msk,
                               const float* __restrict__ st, const float* __restrict__ en,
                               const float* __restrict__ tr,
                               float* __restrict__ res, float* __restrict__ msum) {
  __shared__ float sc[32];
  __shared__ float trs[TT * TT];
  int b = blockIdx.x, j = threadIdx.x;
  for (int i = j; i < TT * TT; i += 32) trs[i] = tr[i];
  if (j < TT) sc[j] = st[j] + em[(size_t)(b * SS) * TT + j];
  __syncthreads();
  for (int s = 1; s < SS; ++s) {
    float nv = 0.f;
    if (j < TT) {
      float mx = -1e30f;
      for (int i = 0; i < TT; ++i) mx = fmaxf(mx, sc[i] + trs[i * TT + j]);
      float sum = 0.f;
      for (int i = 0; i < TT; ++i) sum += expf(sc[i] + trs[i * TT + j] - mx);
      nv = mx + logf(sum) + em[(size_t)(b * SS + s) * TT + j];
      if (!msk[b * SS + s]) nv = sc[j];
    }
    __syncthreads();
    if (j < TT) sc[j] = nv;
    __syncthreads();
  }
  if (j == 0) {
    float mx = -1e30f;
    for (int i = 0; i < TT; ++i) mx = fmaxf(mx, sc[i] + en[i]);
    float sum = 0.f;
    for (int i = 0; i < TT; ++i) sum += expf(sc[i] + en[i] - mx);
    float logZ = mx + logf(sum);
    int t0 = tags[b * SS];
    float num = st[t0] + em[(size_t)(b * SS) * TT + t0];
    int prev = t0;
    float ms = msk[b * SS] ? 1.f : 0.f;
    int seqlen = msk[b * SS] ? 1 : 0;
    for (int s = 1; s < SS; ++s) {
      int tg = tags[b * SS + s];
      float mm = msk[b * SS + s] ? 1.f : 0.f;
      num += mm * (trs[prev * TT + tg] + em[(size_t)(b * SS + s) * TT + tg]);
      prev = tg; ms += mm; seqlen += msk[b * SS + s] ? 1 : 0;
    }
    int lastTag = tags[b * SS + (seqlen > 0 ? seqlen - 1 : 0)];
    num += en[lastTag];
    res[b] = num - logZ;
    msum[b] = ms;
  }
}

__global__ void finalize_kernel(const float* __restrict__ res, const float* __restrict__ msum,
                                float* __restrict__ loss) {
  __shared__ float r[32], m[32];
  int j = threadIdx.x;
  r[j] = res[j]; m[j] = msum[j];
  __syncthreads();
  if (j == 0) {
    float sr = 0.f, sm = 0.f;
    for (int i = 0; i < 32; ++i) { sr += r[i]; sm += m[i]; }
    *loss = -sr / sm;
  }
}

extern "C" void kernel_launch(void* const* d_in, const int* in_sizes, int n_in,
                              void* d_out, int out_size, void* d_ws, size_t ws_size,
                              hipStream_t stream) {
  (void)in_sizes; (void)n_in; (void)out_size; (void)ws_size;
  const int*   xw    = (const int*)  d_in[0];
  const float* xp    = (const float*)d_in[1];
  const int*   xc    = (const int*)  d_in[2];
  const float* xe    = (const float*)d_in[3];
  const unsigned char* msk = (const unsigned char*)d_in[4];
  const int*   yw    = (const int*)  d_in[5];
  const float* wemb  = (const float*)d_in[6];
  const float* cemb  = (const float*)d_in[7];
  const float* cnnW  = (const float*)d_in[8];
  const float* cnnB  = (const float*)d_in[9];
  const float* lin1W = (const float*)d_in[10];
  const float* lin1B = (const float*)d_in[11];
  const float* lin2W = (const float*)d_in[12];
  const float* lin2B = (const float*)d_in[13];
  const float* crfS  = (const float*)d_in[14];
  const float* crfE  = (const float*)d_in[15];
  const float* crfT  = (const float*)d_in[16];

  // workspace carve-out (256B aligned)
  char* ws = (char*)d_ws;
  size_t off = 0;
  auto carve = [&](size_t bytes) -> char* {
    char* p = ws + off;
    off = (off + bytes + 255) & ~(size_t)255;
    return p;
  };
  float*  cpool = (float*) carve((size_t)NROWS * CNNO * 4);
  __bf16* x0    = (__bf16*)carve((size_t)NROWS * IN0P * 2);
  __bf16* wc0f  = (__bf16*)carve((size_t)FOURH * K0TOT * 2);
  __bf16* wc0b  = (__bf16*)carve((size_t)FOURH * K0TOT * 2);
  __bf16* wc1f  = (__bf16*)carve((size_t)FOURH * K1TOT * 2);
  __bf16* wc1b  = (__bf16*)carve((size_t)FOURH * K1TOT * 2);
  __bf16* h0    = (__bf16*)carve((size_t)NROWS * IN1D * 2);
  __bf16* h1    = (__bf16*)carve((size_t)NROWS * IN1D * 2);
  __bf16* l1wb  = (__bf16*)carve((size_t)128 * 1024 * 2);
  __bf16* l2wb  = (__bf16*)carve((size_t)32 * 128 * 2);
  __bf16* h1o   = (__bf16*)carve((size_t)NROWS * 128 * 2);
  int*    hist  = (int*)   carve((size_t)BB * SS * 32 * 4);
  float*  res   = (float*) carve(BB * 4);
  float*  msum  = (float*) carve(BB * 4);

  float* em_out   = (float*)d_out;                       // (32,256,18)
  float* dec_out  = em_out + (size_t)NROWS * TT;         // (32,256)
  float* loss_out = dec_out + (size_t)NROWS;             // scalar

  // 1. weight prep (bf16)
  {
    int tot0 = FOURH * K0TOT, tot1 = FOURH * K1TOT;
    pack_wcat_kernel<<<(tot0 + 255) / 256, 256, 0, stream>>>((const float*)d_in[17], (const float*)d_in[18], wc0f, IN0D, IN0P, K0TOT);
    pack_wcat_kernel<<<(tot0 + 255) / 256, 256, 0, stream>>>((const float*)d_in[20], (const float*)d_in[21], wc0b, IN0D, IN0P, K0TOT);
    pack_wcat_kernel<<<(tot1 + 255) / 256, 256, 0, stream>>>((const float*)d_in[23], (const float*)d_in[24], wc1f, IN1D, IN1D, K1TOT);
    pack_wcat_kernel<<<(tot1 + 255) / 256, 256, 0, stream>>>((const float*)d_in[26], (const float*)d_in[27], wc1b, IN1D, IN1D, K1TOT);
    cvt_rows_kernel<<<(128 * 1024 + 255) / 256, 256, 0, stream>>>(lin1W, l1wb, 128, 128, 1024);
    cvt_rows_kernel<<<(32 * 128 + 255) / 256, 256, 0, stream>>>(lin2W, l2wb, TT, 32, 128);
  }

  // 2. char CNN + input build
  char_cnn_kernel<<<NROWS, 128, 0, stream>>>(xc, cemb, cnnW, cnnB, cpool);
  build_x_kernel<<<NROWS, 128, 0, stream>>>(xw, xp, xe, wemb, cpool, x0);

  // 3. BiLSTM layer 0 (persistent, one WGP per direction)
  {
    size_t shm = (2 * (size_t)BB * IN0P + (size_t)BB * HH + (size_t)BB * FOURH) * 2;  // 208896 B
    lstm_layer_kernel<<<2, 1024, shm, stream>>>(x0, IN0P, IN0P, K0TOT, wc0f, wc0b,
                                                (const float*)d_in[19], (const float*)d_in[22], h0);
  }
  // 4. BiLSTM layer 1
  {
    size_t shm = (2 * (size_t)BB * IN1D + (size_t)BB * HH + (size_t)BB * FOURH) * 2;  // 294912 B
    lstm_layer_kernel<<<2, 1024, shm, stream>>>(h0, IN1D, IN1D, K1TOT, wc1f, wc1b,
                                                (const float*)d_in[25], (const float*)d_in[28], h1);
  }

  // 5. linear heads -> emissions (f32 into d_out)
  gemm_bias_kernel<<<dim3(NROWS / 16, 128 / 16), 32, 0, stream>>>(h1, 1024, l1wb, lin1B, 128, nullptr, h1o, 128);
  gemm_bias_kernel<<<dim3(NROWS / 16, 32 / 16), 32, 0, stream>>>(h1o, 128, l2wb, lin2B, TT, em_out, nullptr, TT);

  // 6. CRF decode + loss
  viterbi_kernel<<<BB, 32, 0, stream>>>(em_out, crfS, crfE, crfT, hist, dec_out);
  crf_nll_kernel<<<BB, 32, 0, stream>>>(em_out, yw, msk, crfS, crfE, crfT, res, msum);
  finalize_kernel<<<1, 32, 0, stream>>>(res, msum, loss_out);
}
